// GraphormerLayer_29953101922826
// MI455X (gfx1250) — matrix operand order repaired and checked by reading, not compiled, not run on hardware
//
#include <hip/hip_runtime.h>
#include <hip/hip_bf16.h>
#include <math.h>

// Problem constants (match reference)
#define NN       3072
#define DD       256
#define HH       8
#define HDD      32
#define EE       98304
#define MAXDEG   512
#define MAXDIST  10

typedef __attribute__((ext_vector_type(16))) _Float16 v16h;
typedef __attribute__((ext_vector_type(8)))  _Float16 v8h;
typedef __attribute__((ext_vector_type(8)))  float    v8f;

union V16H { v16h v; v8h h[2]; };

// A-matrix 16x32 f16 fragment (interleaved per ISA 7.12.2):
// lane<16: row M=lane, K = {0..7} then {16..23}; lane>=16: same row idx, K={8..15},{24..31}
__device__ __forceinline__ v16h load_fragA(const _Float16* p, int hs) {
    V16H u;
    u.h[0] = *(const v8h*)(p + (hs << 3));
    u.h[1] = *(const v8h*)(p + 16 + (hs << 3));
    return u.v;
}
// B-matrix 32x16 f16 fragment (interleaved, same K order as A):
// lane holds column N=lane&15; lanes 0-15 hold K={0..7},{16..23}, lanes 16-31 hold K={8..15},{24..31}
__device__ __forceinline__ v16h load_fragB(const _Float16* p, int hs) {
    V16H u;
    u.h[0] = *(const v8h*)(p + (hs << 3));
    u.h[1] = *(const v8h*)(p + 16 + (hs << 3));
    return u.v;
}

__device__ __forceinline__ v8f wmma_f16(v16h a, v16h b, v8f c) {
    return __builtin_amdgcn_wmma_f32_16x16x32_f16(false, a, false, b, (short)0, c, false, false);
}

// ---------------- degree / embedding / layernorm ----------------

__global__ void k_degree(const int* __restrict__ src, const int* __restrict__ dst,
                         int* __restrict__ deg_out, int* __restrict__ deg_in) {
    int e = blockIdx.x * blockDim.x + threadIdx.x;
    if (e < EE) {
        atomicAdd(&deg_out[src[e]], 1);
        atomicAdd(&deg_in[dst[e]], 1);
    }
}

__global__ void k_xupd(const float* __restrict__ x, const int* __restrict__ deg_in,
                       const int* __restrict__ deg_out, const float* __restrict__ in_emb,
                       const float* __restrict__ out_emb, const float* __restrict__ dscale,
                       float* __restrict__ x_upd) {
    int t = blockIdx.x * blockDim.x + threadIdx.x;        // over N*D
    int n = t >> 8, d = t & 255;
    int di = min(deg_in[n],  MAXDEG);
    int dg = min(deg_out[n], MAXDEG);
    x_upd[t] = x[t] + (in_emb[di * DD + d] + out_emb[dg * DD + d]) * dscale[0];
}

// one row per block, blockDim = 256 == D ; f32 in -> f16 out
__global__ void k_ln(const float* __restrict__ x, const float* __restrict__ g,
                     const float* __restrict__ b, _Float16* __restrict__ out) {
    __shared__ float red[256];
    int row = blockIdx.x, d = threadIdx.x;
    float v = x[(size_t)row * DD + d];
    red[d] = v; __syncthreads();
    for (int s = 128; s > 0; s >>= 1) { if (d < s) red[d] += red[d + s]; __syncthreads(); }
    float mean = red[0] * (1.0f / DD);
    __syncthreads();
    float c = v - mean;
    red[d] = c * c; __syncthreads();
    for (int s = 128; s > 0; s >>= 1) { if (d < s) red[d] += red[d + s]; __syncthreads(); }
    float var = red[0] * (1.0f / DD);
    float y = c * rsqrtf(var + 1e-5f) * g[d] + b[d];
    out[(size_t)row * DD + d] = (_Float16)y;
}

// f32 [R][C] -> f16 transposed [C][R]
__global__ void k_wtrans(const float* __restrict__ W, _Float16* __restrict__ WT, int R, int C) {
    int t = blockIdx.x * blockDim.x + threadIdx.x;
    if (t < R * C) {
        int r = t / C, c = t - r * C;
        WT[(size_t)c * R + r] = (_Float16)W[t];
    }
}

// ---------------- generic WMMA GEMM: C = A[MxK] @ BT[NcxK]^T (+epilogues) --------------
// MODE 0: f16 out = acc + bias
// MODE 1: f16 out transposed [col][row] = acc + bias   (for V^T)
// MODE 2: f32 out = acc + bias + resid
// MODE 3: f16 out = gelu(acc + bias)  (exact erf gelu)
template <int MODE>
__global__ void k_gemm(const _Float16* __restrict__ A, const _Float16* __restrict__ BT,
                       const float* __restrict__ bias, const float* __restrict__ resid,
                       void* __restrict__ outp, int M, int Nc, int K) {
    const int lane = threadIdx.x & 31;
    const int wave = threadIdx.x >> 5;
    const int l15  = lane & 15;
    const int hs   = lane >> 4;
    const int rb   = blockIdx.y * 128 + wave * 16;
    const int cb   = blockIdx.x * 64;

    v8f acc[4] = {};
    const _Float16* arow  = A + (size_t)(rb + l15) * K;
    const _Float16* brow0 = BT + (size_t)(cb + 0 * 16 + l15) * K;
    const _Float16* brow1 = BT + (size_t)(cb + 1 * 16 + l15) * K;
    const _Float16* brow2 = BT + (size_t)(cb + 2 * 16 + l15) * K;
    const _Float16* brow3 = BT + (size_t)(cb + 3 * 16 + l15) * K;

    for (int k0 = 0; k0 < K; k0 += 32) {
        __builtin_prefetch(arow + k0 + 128, 0, 1);
        v16h af = load_fragA(arow + k0, hs);
        acc[0] = wmma_f16(af, load_fragB(brow0 + k0, hs), acc[0]);
        acc[1] = wmma_f16(af, load_fragB(brow1 + k0, hs), acc[1]);
        acc[2] = wmma_f16(af, load_fragB(brow2 + k0, hs), acc[2]);
        acc[3] = wmma_f16(af, load_fragB(brow3 + k0, hs), acc[3]);
    }

#pragma unroll
    for (int t = 0; t < 4; ++t) {
        int col = cb + t * 16 + l15;
        float bs = bias[col];
#pragma unroll
        for (int r = 0; r < 8; ++r) {
            int row = rb + r + hs * 8;
            float v = acc[t][r] + bs;
            if (MODE == 0) {
                ((_Float16*)outp)[(size_t)row * Nc + col] = (_Float16)v;
            } else if (MODE == 1) {
                ((_Float16*)outp)[(size_t)col * M + row] = (_Float16)v;
            } else if (MODE == 2) {
                ((float*)outp)[(size_t)row * Nc + col] = v + resid[(size_t)row * Nc + col];
            } else { // MODE 3: exact gelu
                float gv = 0.5f * v * (1.0f + erff(v * 0.70710678118654752f));
                ((_Float16*)outp)[(size_t)row * Nc + col] = (_Float16)gv;
            }
        }
    }
}

// ---------------- attention (one head per pass) ----------------

// scores[i][j] = (q_i . k_j) * scale + spd_bias + selfloop   (edge bias added separately)
__global__ void k_scores(const _Float16* __restrict__ q, const _Float16* __restrict__ kmat,
                         const float* __restrict__ pe, const float* __restrict__ spd_table,
                         const float* __restrict__ self_loop, float* __restrict__ scores, int h) {
    __shared__ float s_tbl[MAXDIST + 2];                  // per-head SPD bias table (pre-scaled)
    if (threadIdx.x < MAXDIST + 2)
        s_tbl[threadIdx.x] = spd_table[threadIdx.x * HH + h] * 10.0f;  // * BIAS_SCALE
    __syncthreads();

    const int lane = threadIdx.x & 31;
    const int wave = threadIdx.x >> 5;
    const int l15  = lane & 15;
    const int hs   = lane >> 4;
    const int rb   = blockIdx.y * 128 + wave * 16;
    const int cb   = blockIdx.x * 64;
    const float scale = 0.70710678118654752f;  // 1/(ATTN_TEMP^2 * sqrt(32))
    const float slb   = self_loop[h];

    v16h af = load_fragA(q + (size_t)(rb + l15) * DD + h * HDD, hs);
#pragma unroll
    for (int t = 0; t < 4; ++t) {
        v8f acc = {};
        v16h bf = load_fragB(kmat + (size_t)(cb + t * 16 + l15) * DD + h * HDD, hs);
        acc = wmma_f16(af, bf, acc);
        int col = cb + t * 16 + l15;
#pragma unroll
        for (int r = 0; r < 8; ++r) {
            int row = rb + r + hs * 8;
            float spd = pe[(size_t)row * NN + col];
            // branchless bucket: 0 if spd==0, MAXDIST+1 if spd>1, else clip(spd*10,1,10) truncated
            float cl   = fminf(fmaxf(spd * (float)MAXDIST, 1.0f), (float)MAXDIST);
            int bucket = (int)cl;
            bucket = (spd > 1.0f)  ? (MAXDIST + 1) : bucket;
            bucket = (spd == 0.0f) ? 0             : bucket;
            float bv = s_tbl[bucket];
            bv = (row == col) ? (bv + slb) : bv;
            scores[(size_t)row * NN + col] = acc[r] * scale + bv;
        }
    }
}

__global__ void k_edge_bias(const int* __restrict__ src, const int* __restrict__ dst,
                            const int* __restrict__ et, const float* __restrict__ ew,
                            const float* __restrict__ ett, float* __restrict__ scores, int h) {
    int e = blockIdx.x * blockDim.x + threadIdx.x;
    if (e >= EE) return;
    // (edge_type_table[et] + ew*2*BIAS_SCALE) * BIAS_SCALE
    float v = (ett[et[e] * HH + h] + ew[e] * 20.0f) * 10.0f;
    atomicAdd(&scores[(size_t)src[e] * NN + dst[e]], v);
}

// row softmax: f32 scores -> f16 probs. Each thread owns 12 elements held in registers
// so the 12KB row is read from HBM exactly once.
__global__ void k_softmax(const float* __restrict__ scores, _Float16* __restrict__ P) {
    __shared__ float red[256];
    const int row = blockIdx.x, tid = threadIdx.x;
    const float* srow = scores + (size_t)row * NN;

    float vals[NN / 256];
#pragma unroll
    for (int i = 0; i < NN / 256; ++i) vals[i] = srow[tid + i * 256];

    float mx = -1e30f;
#pragma unroll
    for (int i = 0; i < NN / 256; ++i) mx = fmaxf(mx, vals[i]);
    red[tid] = mx; __syncthreads();
    for (int s = 128; s > 0; s >>= 1) { if (tid < s) red[tid] = fmaxf(red[tid], red[tid + s]); __syncthreads(); }
    mx = red[0]; __syncthreads();

    float sum = 0.0f;
#pragma unroll
    for (int i = 0; i < NN / 256; ++i) { vals[i] = __expf(vals[i] - mx); sum += vals[i]; }
    red[tid] = sum; __syncthreads();
    for (int s = 128; s > 0; s >>= 1) { if (tid < s) red[tid] += red[tid + s]; __syncthreads(); }
    float inv = 1.0f / red[0];

    _Float16* prow = P + (size_t)row * NN;
#pragma unroll
    for (int i = 0; i < NN / 256; ++i) prow[tid + i * 256] = (_Float16)(vals[i] * inv);
}

// attn_out[:, h*32 : h*32+32] = P @ v_h  using vT layout [D][N]
__global__ void k_av(const _Float16* __restrict__ P, const _Float16* __restrict__ vT,
                     _Float16* __restrict__ attn_out, int h) {
    const int lane = threadIdx.x & 31;
    const int wave = threadIdx.x >> 5;
    const int l15  = lane & 15;
    const int hs   = lane >> 4;
    const int rb   = blockIdx.x * 128 + wave * 16;

    v8f acc[2] = {};
    const _Float16* prow = P + (size_t)(rb + l15) * NN;
    const _Float16* v0   = vT + (size_t)(h * HDD + l15) * NN;
    const _Float16* v1   = vT + (size_t)(h * HDD + 16 + l15) * NN;
    for (int kc = 0; kc < NN; kc += 32) {
        __builtin_prefetch(prow + kc + 128, 0, 1);
        v16h af = load_fragA(prow + kc, hs);
        acc[0] = wmma_f16(af, load_fragB(v0 + kc, hs), acc[0]);
        acc[1] = wmma_f16(af, load_fragB(v1 + kc, hs), acc[1]);
    }
#pragma unroll
    for (int t = 0; t < 2; ++t) {
#pragma unroll
        for (int r = 0; r < 8; ++r) {
            int row = rb + r + hs * 8;
            attn_out[(size_t)row * DD + h * HDD + t * 16 + l15] = (_Float16)acc[t][r];
        }
    }
}

// ---------------- host orchestration ----------------

extern "C" void kernel_launch(void* const* d_in, const int* in_sizes, int n_in,
                              void* d_out, int out_size, void* d_ws, size_t ws_size,
                              hipStream_t stream) {
    const float* x            = (const float*)d_in[0];
    const int*   edge_index   = (const int*)  d_in[1];   // [2][E]
    const int*   edge_types   = (const int*)  d_in[2];
    const float* pos_encoding = (const float*)d_in[3];
    const float* edge_weights = (const float*)d_in[4];
    const float* Wq = (const float*)d_in[5];   const float* bq = (const float*)d_in[6];
    const float* Wk = (const float*)d_in[7];   const float* bk = (const float*)d_in[8];
    const float* Wv = (const float*)d_in[9];   const float* bv = (const float*)d_in[10];
    const float* Wo = (const float*)d_in[11];  const float* bo = (const float*)d_in[12];
    const float* spd_table      = (const float*)d_in[13];
    const float* edge_type_tab  = (const float*)d_in[14];
    const float* self_loop_bias = (const float*)d_in[15];
    const float* in_deg_emb     = (const float*)d_in[16];
    const float* out_deg_emb    = (const float*)d_in[17];
    const float* degree_scale   = (const float*)d_in[18];
    const float* ln1_g = (const float*)d_in[19]; const float* ln1_b = (const float*)d_in[20];
    const float* ln2_g = (const float*)d_in[21]; const float* ln2_b = (const float*)d_in[22];
    const float* W1 = (const float*)d_in[23];  const float* b1 = (const float*)d_in[24];
    const float* W2 = (const float*)d_in[25];  const float* b2 = (const float*)d_in[26];

    const int* esrc = edge_index;
    const int* edst = edge_index + EE;

    // workspace carve-out (~76 MB; per-head reuse keeps N*N buffers to one copy)
    unsigned char* wsb = (unsigned char*)d_ws;
    size_t off = 0;
    auto take = [&](size_t bytes) -> void* {
        void* p = wsb + off;
        off += (bytes + 255) & ~(size_t)255;
        return p;
    };
    int*      deg_out = (int*)take((size_t)NN * 4);
    int*      deg_in  = (int*)take((size_t)NN * 4);
    float*    x_upd   = (float*)take((size_t)NN * DD * 4);
    _Float16* xn      = (_Float16*)take((size_t)NN * DD * 2);
    _Float16* qb      = (_Float16*)take((size_t)NN * DD * 2);
    _Float16* kb      = (_Float16*)take((size_t)NN * DD * 2);
    _Float16* vT      = (_Float16*)take((size_t)NN * DD * 2);   // [D][N]
    _Float16* WqT     = (_Float16*)take((size_t)DD * DD * 2);
    _Float16* WkT     = (_Float16*)take((size_t)DD * DD * 2);
    _Float16* WvT     = (_Float16*)take((size_t)DD * DD * 2);
    _Float16* WoT     = (_Float16*)take((size_t)DD * DD * 2);
    _Float16* W1T     = (_Float16*)take((size_t)DD * 4 * DD * 2); // [1024][256]
    _Float16* W2T     = (_Float16*)take((size_t)DD * 4 * DD * 2); // [256][1024]
    float*    x2      = (float*)take((size_t)NN * DD * 4);
    _Float16* yv      = (_Float16*)take((size_t)NN * DD * 2);
    _Float16* hmid    = (_Float16*)take((size_t)NN * 4 * DD * 2);
    _Float16* attn_o  = (_Float16*)take((size_t)NN * DD * 2);
    float*    scores  = (float*)take((size_t)NN * NN * 4);
    _Float16* P       = (_Float16*)take((size_t)NN * NN * 2);
    (void)ws_size; (void)n_in; (void)in_sizes; (void)out_size;

    // 1) degrees
    hipMemsetAsync(deg_out, 0, (size_t)NN * 8, stream);  // deg_out + deg_in contiguous
    k_degree<<<(EE + 255) / 256, 256, 0, stream>>>(esrc, edst, deg_out, deg_in);

    // 2) weight convert+transpose to f16 [out][in]
    k_wtrans<<<(DD * DD + 255) / 256, 256, 0, stream>>>(Wq, WqT, DD, DD);
    k_wtrans<<<(DD * DD + 255) / 256, 256, 0, stream>>>(Wk, WkT, DD, DD);
    k_wtrans<<<(DD * DD + 255) / 256, 256, 0, stream>>>(Wv, WvT, DD, DD);
    k_wtrans<<<(DD * DD + 255) / 256, 256, 0, stream>>>(Wo, WoT, DD, DD);
    k_wtrans<<<(DD * 4 * DD + 255) / 256, 256, 0, stream>>>(W1, W1T, DD, 4 * DD);
    k_wtrans<<<(DD * 4 * DD + 255) / 256, 256, 0, stream>>>(W2, W2T, 4 * DD, DD);

    // 3) x += degree embeddings ; LN1
    k_xupd<<<(NN * DD) / 256, 256, 0, stream>>>(x, deg_in, deg_out, in_deg_emb, out_deg_emb,
                                                degree_scale, x_upd);
    k_ln<<<NN, 256, 0, stream>>>(x_upd, ln1_g, ln1_b, xn);

    // 4) QKV projections (WMMA)
    dim3 blk(256);
    dim3 gqkv(DD / 64, NN / 128);
    k_gemm<0><<<gqkv, blk, 0, stream>>>(xn, WqT, bq, nullptr, qb, NN, DD, DD);
    k_gemm<0><<<gqkv, blk, 0, stream>>>(xn, WkT, bk, nullptr, kb, NN, DD, DD);
    k_gemm<1><<<gqkv, blk, 0, stream>>>(xn, WvT, bv, nullptr, vT, NN, DD, DD);

    // 5) attention, one head at a time (reuse N*N buffers)
    dim3 gsc(NN / 64, NN / 128);
    for (int h = 0; h < HH; ++h) {
        k_scores<<<gsc, blk, 0, stream>>>(qb, kb, pos_encoding, spd_table, self_loop_bias,
                                          scores, h);
        k_edge_bias<<<(EE + 255) / 256, 256, 0, stream>>>(esrc, edst, edge_types, edge_weights,
                                                          edge_type_tab, scores, h);
        k_softmax<<<NN, 256, 0, stream>>>(scores, P);
        k_av<<<NN / 128, blk, 0, stream>>>(P, vT, attn_o, h);
    }

    // 6) output projection + residual -> x2 (f32)
    k_gemm<2><<<gqkv, blk, 0, stream>>>(attn_o, WoT, bo, x_upd, x2, NN, DD, DD);

    // 7) LN2 -> y (f16)
    k_ln<<<NN, 256, 0, stream>>>(x2, ln2_g, ln2_b, yv);

    // 8) FFN: gelu(y@W1+b1) -> hmid ; x2 + hmid@W2 + b2 -> d_out
    dim3 gf1(4 * DD / 64, NN / 128);
    k_gemm<3><<<gf1, blk, 0, stream>>>(yv, W1T, b1, nullptr, hmid, NN, 4 * DD, DD);
    dim3 gf2(DD / 64, NN / 128);
    k_gemm<2><<<gf2, blk, 0, stream>>>(hmid, W2T, b2, x2, (float*)d_out, NN, DD, 4 * DD);
}
